// BasicTransformerBlock_80427557585106
// MI455X (gfx1250) — matrix operand order repaired and checked by reading, not compiled, not run on hardware
//
#include <hip/hip_runtime.h>
#include <cstdint>
#include <cstddef>

// ---------------------------------------------------------------------------
// MI455X / gfx1250 implementation of BasicTransformerBlock.
// Heavy math: V_WMMA_F32_16X16X32_F16 (f16 in, f32 accum).
// GEMM B-panels are staged global->LDS with GLOBAL_LOAD_ASYNC_TO_LDS_B128
// (ASYNCcnt) into a double buffer; weights are pre-transposed to f16 so the
// async copies need no data reshaping.
// ---------------------------------------------------------------------------

typedef _Float16 half_t;
typedef __attribute__((ext_vector_type(8)))  _Float16 v8h;
typedef __attribute__((ext_vector_type(16))) _Float16 v16h;
typedef __attribute__((ext_vector_type(8)))  float    v8f;

__device__ inline v8f wmma_f16(v16h a, v16h b, v8f c) {
  return __builtin_amdgcn_wmma_f32_16x16x32_f16(false, a, false, b, (short)0, c,
                                                false, false);
}
__device__ inline v8f  zero8f()  { v8f z;  for (int i = 0; i < 8;  ++i) z[i] = 0.f;         return z; }
__device__ inline v8h  ld8 (const half_t* p) { return *(const v8h*)p;  }
__device__ inline v16h ld16(const half_t* p) { return *(const v16h*)p; }
__device__ inline v16h cat(v8h lo, v8h hi) {
  union { v16h v; v8h h[2]; } u; u.h[0] = lo; u.h[1] = hi; return u.v;
}
__device__ inline float rmax16(float v) { for (int m = 1; m < 16; m <<= 1) v = fmaxf(v, __shfl_xor(v, m, 32)); return v; }
__device__ inline float rsum16(float v) { for (int m = 1; m < 16; m <<= 1) v += __shfl_xor(v, m, 32);          return v; }
__device__ inline float rmax32(float v) { for (int m = 1; m < 32; m <<= 1) v = fmaxf(v, __shfl_xor(v, m, 32)); return v; }
__device__ inline float rsum32(float v) { for (int m = 1; m < 32; m <<= 1) v += __shfl_xor(v, m, 32);          return v; }

// issue one 16B global->LDS async copy (tracked by ASYNCcnt)
__device__ inline void async_cp16(uint32_t lds_off, const half_t* gsrc) {
  const unsigned long long g = (unsigned long long)(uintptr_t)gsrc;
  asm volatile("global_load_async_to_lds_b128 %0, %1, off"
               :: "v"(lds_off), "v"(g) : "memory");
}
__device__ inline void wait_async0() {
  asm volatile("s_wait_asynccnt 0x0" ::: "memory");
}
__device__ inline void wait_ds0() {
  asm volatile("s_wait_dscnt 0x0" ::: "memory");
}

// ---------------------------------------------------------------------------
// fp32 -> f16 convert (plain, for A-side activations / context)
// ---------------------------------------------------------------------------
__global__ void cvt_f16(const float* __restrict__ s, half_t* __restrict__ d, int n) {
  const int i = blockIdx.x * 256 + threadIdx.x;
  if (i < n) d[i] = (half_t)s[i];
}

// fp32 W[K][N] -> f16 Wt[N][K] (one-time; makes B-panels async-copyable)
__global__ void cvt_f16_t(const float* __restrict__ s, half_t* __restrict__ d,
                          int K, int N) {
  const int i = blockIdx.x * 256 + threadIdx.x;
  if (i < K * N) {
    const int k = i / N, n = i - k * N;
    d[(size_t)n * K + k] = (half_t)s[i];
  }
}

// ---------------------------------------------------------------------------
// LayerNorm over 320 cols, one wave per row, f16 output for WMMA consumers
// ---------------------------------------------------------------------------
__global__ void ln_f16(const float* __restrict__ x, const float* __restrict__ g,
                       const float* __restrict__ b, half_t* __restrict__ o, int rows) {
  const int lane = threadIdx.x & 31, w = threadIdx.x >> 5;
  const int row = blockIdx.x * 4 + w;
  if (row >= rows) return;
  const float* xr = x + (size_t)row * 320;
  float s = 0.f, s2 = 0.f;
  for (int i = lane; i < 320; i += 32) { float v = xr[i]; s += v; s2 += v * v; }
  s = rsum32(s); s2 = rsum32(s2);
  const float mu  = s * (1.f / 320.f);
  const float var = s2 * (1.f / 320.f) - mu * mu;
  const float rs  = rsqrtf(var + 1e-5f);
  half_t* orow = o + (size_t)row * 320;
  for (int i = lane; i < 320; i += 32)
    orow[i] = (half_t)((xr[i] - mu) * rs * g[i] + b[i]);
}

// ---------------------------------------------------------------------------
// WMMA GEMM: C[M,N] = A[M,K]f16 * Bt[N,K]f16^T (+bias)(+residual)
// Block = 128 threads (4 waves), block tile 64x64, K step 32.
// B panel (64 cols x 32 k) async-staged into a double-buffered LDS tile:
// one barrier per K-step, copies overlap the WMMAs.
// ---------------------------------------------------------------------------
__global__ void gemm_f16(const half_t* __restrict__ A, const half_t* __restrict__ Bt,
                         const float* __restrict__ bias, const float* __restrict__ res,
                         float* __restrict__ Cf, half_t* __restrict__ Ch,
                         int M, int N, int K) {
  __shared__ __align__(32) half_t Bs[2][64][32];  // Bs[buf][n][k]
  const int t = threadIdx.x;
  const int lane = t & 31, wave = t >> 5;
  const int grp = lane >> 4, l15 = lane & 15, b8 = grp * 8;
  const int bm = blockIdx.x * 64, bn = blockIdx.y * 64;

  int ar = bm + wave * 16 + l15;        // A row this lane feeds
  if (ar >= M) ar = M - 1;              // clamp; stores are guarded below
  const half_t* ap = A + (size_t)ar * K;

  // staging assignment: thread t copies 32B of row (bn + t/2), halves sc..sc+15
  const int sr = t >> 1, sc = (t & 1) * 16;
  const half_t* srow = Bt + (size_t)(bn + sr) * K + sc;
  const uint32_t d0 = (uint32_t)(uintptr_t)&Bs[0][sr][sc];
  const uint32_t d1 = (uint32_t)(uintptr_t)&Bs[1][sr][sc];

  v8f acc[4]; for (int i = 0; i < 4; ++i) acc[i] = zero8f();

  // prologue: stage panel kb=0 into buffer 0
  async_cp16(d0, srow);
  async_cp16(d0 + 16u, srow + 8);
  wait_async0();
  __syncthreads();

  int buf = 0;
  for (int kb = 0; kb < K; kb += 32, buf ^= 1) {
    if (kb + 32 < K) {                   // stage next panel into other buffer
      const uint32_t dn = buf ? d0 : d1;
      async_cp16(dn, srow + kb + 32);
      async_cp16(dn + 16u, srow + kb + 40);
    }

    // A fragment straight from global (contiguous k = 16*grp + s, same order as B)
    v16h afrag = ld16(ap + kb + grp * 16);
    if (kb + 32 < K) __builtin_prefetch(ap + kb + 32, 0, 0);

    // B fragments: contiguous 32B LDS reads (lane = column, K contiguous)
    v16h bf0 = *(const v16h*)&Bs[buf][l15][grp * 16];
    v16h bf1 = *(const v16h*)&Bs[buf][16 + l15][grp * 16];
    v16h bf2 = *(const v16h*)&Bs[buf][32 + l15][grp * 16];
    v16h bf3 = *(const v16h*)&Bs[buf][48 + l15][grp * 16];
    acc[0] = wmma_f16(afrag, bf0, acc[0]);
    acc[1] = wmma_f16(afrag, bf1, acc[1]);
    acc[2] = wmma_f16(afrag, bf2, acc[2]);
    acc[3] = wmma_f16(afrag, bf3, acc[3]);

    wait_async0();     // next panel landed in LDS
    __syncthreads();   // all waves done reading current panel
  }

  // epilogue: C layout -> VGPR g holds row (g + 8*grp), col = lane&15
  for (int ns = 0; ns < 4; ++ns) {
    const int col = bn + ns * 16 + l15;
    for (int g = 0; g < 8; ++g) {
      const int row = bm + wave * 16 + g + 8 * grp;
      if (row < M) {
        float v = acc[ns][g];
        if (bias) v += bias[col];
        if (res)  v += res[(size_t)row * N + col];
        if (Cf) Cf[(size_t)row * N + col] = v;
        if (Ch) Ch[(size_t)row * N + col] = (half_t)v;
      }
    }
  }
}

// ---------------------------------------------------------------------------
// Pack q/k/v [4096][320]f16 into per-head WMMA-friendly layouts:
//   Qp/Kp : [8][4096][64]  (Dh padded 40->64, Q pre-scaled by 40^-0.5)
//   Vt    : [8][64][4096]  (transposed so PV B-fragments are contiguous)
// ---------------------------------------------------------------------------
__global__ void pack_qkv(const half_t* __restrict__ q, const half_t* __restrict__ k,
                         const half_t* __restrict__ v, half_t* __restrict__ Qp,
                         half_t* __restrict__ Kp, half_t* __restrict__ Vt) {
  const int idx = blockIdx.x * 256 + threadIdx.x;       // 8*4096*64
  const int d = idx & 63, n = (idx >> 6) & 4095, h = idx >> 18;
  float qv = 0.f, kv = 0.f, vv = 0.f;
  if (d < 40) {
    const int s = n * 320 + h * 40 + d;
    qv = (float)q[s] * 0.15811388300841898f;            // fold softmax scale
    kv = (float)k[s];
    vv = (float)v[s];
  }
  Qp[idx] = (half_t)qv;
  Kp[idx] = (half_t)kv;
  Vt[((size_t)(h * 64 + d) << 12) + n] = (half_t)vv;
}

// ---------------------------------------------------------------------------
// Self-attention, flash style. One wave per (head, 16-query tile).
// Scores via WMMA (Dh padded to 64 -> 2 K-chunks), running max/sum in f32,
// P routed C-layout -> A-layout through a per-wave LDS tile (s_wait_dscnt;
// per-wave LDS ops are in-order on CDNA5 so no barrier is needed).
// ---------------------------------------------------------------------------
__global__ void attn_self(const half_t* __restrict__ Qp, const half_t* __restrict__ Kp,
                          const half_t* __restrict__ Vt, half_t* __restrict__ out) {
  __shared__ __align__(32) half_t P[4][16][32];
  const int lane = threadIdx.x & 31, w = threadIdx.x >> 5;
  const int wid = blockIdx.x * 4 + w;
  const int h = wid >> 8, tile = wid & 255;
  const int grp = lane >> 4, l15 = lane & 15, b8 = grp * 8;

  const half_t* qrow = Qp + ((size_t)h * 4096 + tile * 16 + l15) * 64;
  const v16h qa0 = ld16(qrow + grp * 16);
  const v16h qa1 = ld16(qrow + 32 + grp * 16);
  const half_t* kbase = Kp + (size_t)h * 4096 * 64;
  const half_t* vbase = Vt + (size_t)h * 64 * 4096;

  v8f o[4]; for (int i = 0; i < 4; ++i) o[i] = zero8f();
  float mr[8], lr[8];
  for (int g = 0; g < 8; ++g) { mr[g] = -1e30f; lr[g] = 0.f; }

  for (int j = 0; j < 4096; j += 32) {
    // S tiles for key cols j..j+15 and j+16..j+31 (B-frag = contiguous K rows)
    const half_t* k0 = kbase + (size_t)(j + l15) * 64 + grp * 16;
    const half_t* k1 = kbase + (size_t)(j + 16 + l15) * 64 + grp * 16;
    v8f s0 = zero8f(), s1 = zero8f();
    s0 = wmma_f16(qa0, ld16(k0), s0);
    s0 = wmma_f16(qa1, ld16(k0 + 32), s0);
    s1 = wmma_f16(qa0, ld16(k1), s1);
    s1 = wmma_f16(qa1, ld16(k1 + 32), s1);

    float p0[8], p1[8];
    for (int g = 0; g < 8; ++g) {
      float mx = rmax16(fmaxf(s0[g], s1[g]));   // row max over 32 cols
      float mn = fmaxf(mr[g], mx);
      float al = __expf(mr[g] - mn);
      mr[g] = mn;
      p0[g] = __expf(s0[g] - mn);
      p1[g] = __expf(s1[g] - mn);
      lr[g] = lr[g] * al + rsum16(p0[g] + p1[g]);
      for (int d = 0; d < 4; ++d) o[d][g] *= al;
    }

    // C-layout P -> LDS -> A-layout fragment
    for (int g = 0; g < 8; ++g) {
      P[w][g + 8 * grp][l15]      = (half_t)p0[g];
      P[w][g + 8 * grp][l15 + 16] = (half_t)p1[g];
    }
    wait_ds0();
    v16h pa = *(const v16h*)&P[w][l15][grp * 16];

    for (int d = 0; d < 4; ++d) {
      const half_t* vp = vbase + (size_t)(d * 16 + l15) * 4096 + j + grp * 16;
      o[d] = wmma_f16(pa, ld16(vp), o[d]);
    }
  }

  for (int ds = 0; ds < 4; ++ds) {
    const int d = ds * 16 + l15;
    if (d < 40) {
      for (int g = 0; g < 8; ++g) {
        const int row = tile * 16 + g + 8 * grp;
        out[(size_t)row * 320 + h * 40 + d] = (half_t)(o[ds][g] / lr[g]);
      }
    }
  }
}

// ---------------------------------------------------------------------------
// Cross-attention (77 keys -> tiny compute): one wave per (head, query row).
// Includes the focused-attention-mask branch. Intra-wave LDS sync via
// s_wait_dscnt (LDS ops are in-order per wave on CDNA5).
// ---------------------------------------------------------------------------
__global__ void attn_cross(const half_t* __restrict__ q2, const half_t* __restrict__ k2,
                           const half_t* __restrict__ v2, const float* __restrict__ fam,
                           const int* __restrict__ use_fca, half_t* __restrict__ out) {
  __shared__ float S[4][80];
  __shared__ float F[4][80];
  const int lane = threadIdx.x & 31, w = threadIdx.x >> 5;
  const int wid = blockIdx.x * 4 + w;
  const int row = wid >> 3, h = wid & 7;
  const half_t* qr = q2 + (size_t)row * 320 + h * 40;

  for (int kk = lane; kk < 77; kk += 32) {
    const half_t* kr = k2 + (size_t)kk * 320 + h * 40;
    float s = 0.f;
    for (int d = 0; d < 40; ++d) s += (float)qr[d] * (float)kr[d];
    S[w][kk] = s * 0.15811388300841898f;
  }
  wait_ds0();

  if (use_fca[0]) {
    for (int d = lane; d < 77; d += 32) {
      float fw = 0.f;
      for (int kk = 0; kk < 77; ++kk) fw += S[w][kk] * fam[d * 77 + kk];
      F[w][d] = fminf(fabsf(fw), 1.f);
    }
    wait_ds0();
    float mx = 0.f;
    for (int d = lane; d < 77; d += 32) mx = fmaxf(mx, F[w][d]);
    mx = rmax32(mx) + 1e-6f;
    for (int d = lane; d < 77; d += 32)
      S[w][d] += (F[w][d] / mx > 0.6f) ? 0.f : -50.f;
    wait_ds0();
  }

  float mloc = -1e30f;
  for (int kk = lane; kk < 77; kk += 32) mloc = fmaxf(mloc, S[w][kk]);
  const float m = rmax32(mloc);
  float sloc = 0.f;
  for (int kk = lane; kk < 77; kk += 32) {
    float p = __expf(S[w][kk] - m);
    S[w][kk] = p;
    sloc += p;
  }
  const float inv = 1.f / rsum32(sloc);
  wait_ds0();

  for (int d = lane; d < 40; d += 32) {
    float o = 0.f;
    for (int kk = 0; kk < 77; ++kk)
      o += S[w][kk] * (float)v2[(size_t)kk * 320 + h * 40 + d];
    out[(size_t)row * 320 + h * 40 + d] = (half_t)(o * inv);
  }
}

// ---------------------------------------------------------------------------
// GEGLU: a * gelu(gate), exact erf formulation
// ---------------------------------------------------------------------------
__global__ void geglu(const half_t* __restrict__ proj, half_t* __restrict__ out) {
  const int idx = blockIdx.x * 256 + threadIdx.x;  // 4096*1280
  const int row = idx / 1280, c = idx - row * 1280;
  const float a  = (float)proj[(size_t)row * 2560 + c];
  const float gt = (float)proj[(size_t)row * 2560 + 1280 + c];
  const float ge = 0.5f * gt * (1.f + erff(gt * 0.7071067811865476f));
  out[idx] = (half_t)(a * ge);
}

// ---------------------------------------------------------------------------
// Orchestration
// ---------------------------------------------------------------------------
extern "C" void kernel_launch(void* const* d_in, const int* in_sizes, int n_in,
                              void* d_out, int out_size, void* d_ws, size_t ws_size,
                              hipStream_t stream) {
  (void)in_sizes; (void)n_in; (void)out_size; (void)ws_size;
  constexpr int N = 4096, DIM = 320, CTX = 768, MC = 77, FF2 = 2560, FF = 1280;

  const float* x    = (const float*)d_in[0];
  const float* ctx  = (const float*)d_in[1];
  const float* fam  = (const float*)d_in[2];
  const float* g1   = (const float*)d_in[3];
  const float* b1   = (const float*)d_in[4];
  const float* g2   = (const float*)d_in[5];
  const float* b2   = (const float*)d_in[6];
  const float* g3   = (const float*)d_in[7];
  const float* b3   = (const float*)d_in[8];
  const float* Wq1  = (const float*)d_in[9];
  const float* Wk1  = (const float*)d_in[10];
  const float* Wv1  = (const float*)d_in[11];
  const float* Wo1  = (const float*)d_in[12];
  const float* bo1  = (const float*)d_in[13];
  const float* Wq2  = (const float*)d_in[14];
  const float* Wk2  = (const float*)d_in[15];
  const float* Wv2  = (const float*)d_in[16];
  const float* Wo2  = (const float*)d_in[17];
  const float* bo2  = (const float*)d_in[18];
  const float* Wff1 = (const float*)d_in[19];
  const float* bff1 = (const float*)d_in[20];
  const float* Wff2 = (const float*)d_in[21];
  const float* bff2 = (const float*)d_in[22];
  const int*   ufca = (const int*)d_in[23];
  float* out = (float*)d_out;

  // bump allocator over workspace
  char* p = (char*)d_ws;
  auto ah = [&](size_t n) -> half_t* { half_t* r = (half_t*)p; p += ((n * 2 + 255) / 256) * 256; return r; };
  auto af = [&](size_t n) -> float*  { float*  r = (float*)p;  p += ((n * 4 + 255) / 256) * 256; return r; };

  half_t* wq1  = ah((size_t)DIM * DIM);   // transposed f16 weights [N][K]
  half_t* wk1  = ah((size_t)DIM * DIM);
  half_t* wv1  = ah((size_t)DIM * DIM);
  half_t* wo1  = ah((size_t)DIM * DIM);
  half_t* wq2  = ah((size_t)DIM * DIM);
  half_t* wk2  = ah((size_t)CTX * DIM);
  half_t* wv2  = ah((size_t)CTX * DIM);
  half_t* wo2  = ah((size_t)DIM * DIM);
  half_t* wff1 = ah((size_t)DIM * FF2);
  half_t* wff2 = ah((size_t)FF * DIM);
  half_t* c16  = ah((size_t)MC * CTX);
  half_t* h1   = ah((size_t)N * DIM);
  half_t* qb   = ah((size_t)N * DIM);
  half_t* kb   = ah((size_t)N * DIM);
  half_t* vb   = ah((size_t)N * DIM);
  half_t* Qp   = ah((size_t)8 * N * 64);
  half_t* Kp   = ah((size_t)8 * N * 64);
  half_t* Vt   = ah((size_t)8 * 64 * N);
  half_t* at1  = ah((size_t)N * DIM);
  float*  x1   = af((size_t)N * DIM);
  half_t* h2   = ah((size_t)N * DIM);
  half_t* q2   = ah((size_t)N * DIM);
  half_t* k2c  = ah((size_t)MC * DIM);
  half_t* v2c  = ah((size_t)MC * DIM);
  half_t* at2  = ah((size_t)N * DIM);
  float*  x2   = af((size_t)N * DIM);
  half_t* h3   = ah((size_t)N * DIM);
  half_t* proj = ah((size_t)N * FF2);
  half_t* ffg  = ah((size_t)N * FF);

  auto cvt = [&](const float* s, half_t* d, int n) {
    cvt_f16<<<dim3((n + 255) / 256), dim3(256), 0, stream>>>(s, d, n);
  };
  auto cvtT = [&](const float* s, half_t* d, int K, int Nc) {   // W[K][N] -> Wt[N][K]
    cvt_f16_t<<<dim3((K * Nc + 255) / 256), dim3(256), 0, stream>>>(s, d, K, Nc);
  };
  auto gemm = [&](const half_t* A, const half_t* Bt, const float* bias, const float* res,
                  float* Cf, half_t* Ch, int M, int Nc, int K) {
    gemm_f16<<<dim3((M + 63) / 64, Nc / 64), dim3(128), 0, stream>>>(A, Bt, bias, res, Cf, Ch, M, Nc, K);
  };
  auto ln = [&](const float* xi, const float* g, const float* b, half_t* o, int rows) {
    ln_f16<<<dim3((rows + 3) / 4), dim3(128), 0, stream>>>(xi, g, b, o, rows);
  };

  // weight conversion (transposed) + context conversion (amortized, tiny)
  cvtT(Wq1, wq1, DIM, DIM);   cvtT(Wk1, wk1, DIM, DIM);
  cvtT(Wv1, wv1, DIM, DIM);   cvtT(Wo1, wo1, DIM, DIM);
  cvtT(Wq2, wq2, DIM, DIM);   cvtT(Wk2, wk2, CTX, DIM);
  cvtT(Wv2, wv2, CTX, DIM);   cvtT(Wo2, wo2, DIM, DIM);
  cvtT(Wff1, wff1, DIM, FF2); cvtT(Wff2, wff2, FF, DIM);
  cvt(ctx, c16, MC * CTX);

  // ---- self-attention ----
  ln(x, g1, b1, h1, N);
  gemm(h1, wq1, nullptr, nullptr, nullptr, qb, N, DIM, DIM);
  gemm(h1, wk1, nullptr, nullptr, nullptr, kb, N, DIM, DIM);
  gemm(h1, wv1, nullptr, nullptr, nullptr, vb, N, DIM, DIM);
  pack_qkv<<<dim3(8 * N * 64 / 256), dim3(256), 0, stream>>>(qb, kb, vb, Qp, Kp, Vt);
  attn_self<<<dim3(8 * 256 / 4), dim3(128), 0, stream>>>(Qp, Kp, Vt, at1);
  gemm(at1, wo1, bo1, x, x1, nullptr, N, DIM, DIM);

  // ---- cross-attention with focused-attention mask ----
  ln(x1, g2, b2, h2, N);
  gemm(h2, wq2, nullptr, nullptr, nullptr, q2, N, DIM, DIM);
  gemm(c16, wk2, nullptr, nullptr, nullptr, k2c, MC, DIM, CTX);
  gemm(c16, wv2, nullptr, nullptr, nullptr, v2c, MC, DIM, CTX);
  attn_cross<<<dim3(8 * N / 4), dim3(128), 0, stream>>>(q2, k2c, v2c, fam, ufca, at2);
  gemm(at2, wo2, bo2, x1, x2, nullptr, N, DIM, DIM);

  // ---- GEGLU FFN ----
  ln(x2, g3, b3, h3, N);
  gemm(h3, wff1, bff1, nullptr, nullptr, proj, N, FF2, DIM);
  geglu<<<dim3(N * FF / 256), dim3(256), 0, stream>>>(proj, ffg);
  gemm(ffg, wff2, bff2, x2, out, nullptr, N, DIM, FF);
}
